// AmalgamatedGNN_27599459844332
// MI455X (gfx1250) — matrix operand order repaired and checked by reading, not compiled, not run on hardware
//
#include <hip/hip_runtime.h>

// ---------------------------------------------------------------------------
// Types for CDNA5 WMMA
// ---------------------------------------------------------------------------
typedef __attribute__((ext_vector_type(16))) __bf16          v16bf;
typedef __attribute__((ext_vector_type(16))) unsigned short  u16x16;
typedef __attribute__((ext_vector_type(8)))  float           v8f;

#define N_NODES 100000
#define N_GRAPHS 512
#define D_IN 128
#define D_H 256
#define D_MID 32
#define N_CLS 2

// gfx1250 async global->LDS (ASYNCcnt path), feature-guarded with fallback
#if defined(__has_builtin)
#  if __has_builtin(__builtin_amdgcn_global_load_async_to_lds_b128)
#    define USE_ASYNC_LDS 1
#  endif
#endif
#ifndef USE_ASYNC_LDS
#  define USE_ASYNC_LDS 0
#endif

#if USE_ASYNC_LDS
// builtin signature (from hipcc diagnostic): (int4 AS1*, int4 AS3*, imm, imm)
typedef int async_v4i __attribute__((vector_size(16)));
typedef __attribute__((address_space(1))) async_v4i* g_v4i_ptr;
typedef __attribute__((address_space(3))) async_v4i* l_v4i_ptr;
#endif

static __device__ __forceinline__ unsigned short f2bf(float f) {
    // round-to-nearest-even f32 -> bf16 (bit trick; NaN edge ignored)
    unsigned int u = __float_as_uint(f);
    u = (u + 0x7FFFu + ((u >> 16) & 1u)) >> 16;
    return (unsigned short)u;
}
static __device__ __forceinline__ float bf2f(unsigned short u) {
    return __uint_as_float((unsigned int)u << 16);
}

// ---------------------------------------------------------------------------
// 0) zero a contiguous region (grid-stride, 16B)
// ---------------------------------------------------------------------------
__global__ void zero_kernel(float4* p, long long n4) {
    long long i = (long long)blockIdx.x * blockDim.x + threadIdx.x;
    long long stride = (long long)gridDim.x * blockDim.x;
    float4 z = make_float4(0.f, 0.f, 0.f, 0.f);
    for (; i < n4; i += stride) p[i] = z;
}

// ---------------------------------------------------------------------------
// 1) pack f32 -> bf16 (flat)
// ---------------------------------------------------------------------------
__global__ void pack_bf16_kernel(const float* __restrict__ in,
                                 unsigned short* __restrict__ out, int n) {
    int i = blockIdx.x * blockDim.x + threadIdx.x;
    if (i < n) out[i] = f2bf(in[i]);
}

// ---------------------------------------------------------------------------
// 2) pack weights [K,256]+[K,256] -> Btg[2*K/32][256][32] bf16
//    (col-major-per-K-tile so each WMMA B fragment is 32 contiguous bytes)
// ---------------------------------------------------------------------------
__global__ void pack_B_kernel(const float* __restrict__ Wl,
                              const float* __restrict__ Wr,
                              int K, unsigned short* __restrict__ out) {
    int half = K >> 5;                       // K-steps per matrix
    int n = (half * 2) * 8192;               // 8192 = 256 cols * 32 k
    int e = blockIdx.x * blockDim.x + threadIdx.x;
    if (e >= n) return;
    int s = e >> 13;
    int r = e & 8191;
    int col = r >> 5, k = r & 31;
    const float* W = (s < half) ? Wl : Wr;
    int ss = (s < half) ? s : s - half;
    out[e] = f2bf(W[(size_t)(ss * 32 + k) * 256 + col]);
}

// ---------------------------------------------------------------------------
// 3) wave-per-edge scatter, f32 features (conv1): agg[dst]+=x[src], deg[dst]+=1
// ---------------------------------------------------------------------------
__global__ void scatter_f32_kernel(const float* __restrict__ feat,   // [N,128]
                                   const int* __restrict__ src,
                                   const int* __restrict__ dst,
                                   float* __restrict__ agg,
                                   float* __restrict__ deg, int nEdges) {
    unsigned int gt = blockIdx.x * blockDim.x + threadIdx.x;
    int e = (int)(gt >> 5);
    int lane = (int)(gt & 31u);
    if (e >= nEdges) return;
    int s = src[e];
    int d = dst[e];
    const float4 v = *(const float4*)(feat + (size_t)s * D_IN + lane * 4);
    float* ad = agg + (size_t)d * D_IN + lane * 4;
    atomicAdd(&ad[0], v.x);
    atomicAdd(&ad[1], v.y);
    atomicAdd(&ad[2], v.z);
    atomicAdd(&ad[3], v.w);
    if (lane == 0) atomicAdd(&deg[d], 1.0f);
}

// ---------------------------------------------------------------------------
// 4) wave-per-edge scatter, bf16 features (conv2): agg[dst] += h[src]
// ---------------------------------------------------------------------------
__global__ void scatter_bf16_kernel(const unsigned short* __restrict__ feat, // [N,256] bf16
                                    const int* __restrict__ src,
                                    const int* __restrict__ dst,
                                    float* __restrict__ agg, int nEdges) {
    unsigned int gt = blockIdx.x * blockDim.x + threadIdx.x;
    int e = (int)(gt >> 5);
    int lane = (int)(gt & 31u);
    if (e >= nEdges) return;
    int s = src[e];
    int d = dst[e];
    const uint4 q = *(const uint4*)(feat + (size_t)s * D_H + lane * 8); // 8 bf16
    float* ad = agg + (size_t)d * D_H + lane * 8;
    unsigned int w[4] = {q.x, q.y, q.z, q.w};
#pragma unroll
    for (int j = 0; j < 4; ++j) {
        atomicAdd(&ad[2 * j + 0], __uint_as_float(w[j] << 16));
        atomicAdd(&ad[2 * j + 1], __uint_as_float(w[j] & 0xFFFF0000u));
    }
}

// ---------------------------------------------------------------------------
// 5) graph sizes: sizes[batch[i]] += 1
// ---------------------------------------------------------------------------
__global__ void sizes_kernel(const int* __restrict__ batch,
                             float* __restrict__ sizes, int n) {
    int i = blockIdx.x * blockDim.x + threadIdx.x;
    if (i < n) atomicAdd(&sizes[batch[i]], 1.0f);
}

// ---------------------------------------------------------------------------
// 6) Fused SAGE GEMM:
//    out = leaky( (A1/max(deg,1)) @ B1 + A2 @ B2 + bias ), B pre-packed bf16
//    mode 0: store bf16 to out_hb [N, 256]
//    mode 1: blended = large ? out : h_prev ; atomicAdd into pooled[batch]
//    Block: 128 threads (4 waves). Block tile: 16 nodes x 256 cols.
// ---------------------------------------------------------------------------
__global__ void __launch_bounds__(128)
sage_gemm_kernel(const float* __restrict__ A1,           // [N, K1] f32 (agg)
                 const unsigned short* __restrict__ A2b, // [N, K2] bf16
                 const unsigned short* __restrict__ Btg, // [(K1+K2)/32][256][32] bf16
                 const float* __restrict__ bias,         // [256]
                 const float* __restrict__ deg,          // [N]
                 unsigned short* __restrict__ out_hb,        // mode 0
                 const unsigned short* __restrict__ h_prevb, // mode 1
                 const int* __restrict__ batch,              // mode 1
                 const float* __restrict__ sizes,            // mode 1
                 float* __restrict__ pooled,                 // mode 1
                 int K1, int K2, int mode) {
    __shared__ __align__(16) unsigned short As[16 * 32];   // [row][k]
    __shared__ __align__(16) unsigned short Bt[256 * 32];  // [col][k]
    __shared__ float rdeg_s[16];

    const int tid  = threadIdx.x;
    const int lane = tid & 31;
    const int wave = tid >> 5;              // 0..3 -> cols wave*64 .. +63
    const int node0 = blockIdx.x * 16;

    if (tid < 16) {
        float d = deg[node0 + tid];
        rdeg_s[tid] = 1.0f / fmaxf(d, 1.0f);
    }

    v8f acc[4] = {};

    const int steps1 = K1 >> 5;
    const int steps2 = K2 >> 5;
    const int row = lane & 15;
    const int hi  = lane >> 4;              // 0 or 1

    for (int s = 0; s < steps1 + steps2; ++s) {
        const bool first = (s < steps1);
        const int k0 = (first ? s : (s - steps1)) << 5;

        __syncthreads();
        // ---- stage A tile (16 x 32): each thread owns 4 contiguous k ----
        {
            int i = tid * 4;
            int r = i >> 5, k = i & 31;
            if (first) {
                const float4 av =
                    *(const float4*)&A1[(size_t)(node0 + r) * K1 + (k0 + k)];
                float sc = rdeg_s[r];
                unsigned int lo = (unsigned int)f2bf(av.x * sc) |
                                  ((unsigned int)f2bf(av.y * sc) << 16);
                unsigned int hiw = (unsigned int)f2bf(av.z * sc) |
                                   ((unsigned int)f2bf(av.w * sc) << 16);
                uint2 u; u.x = lo; u.y = hiw;
                *(uint2*)&As[r * 32 + k] = u;
                if (s + 1 < steps1)   // prefetch next agg K-tile
                    __builtin_prefetch(&A1[(size_t)(node0 + r) * K1 + (k0 + 32 + k)], 0, 0);
            } else {
                *(uint2*)&As[r * 32 + k] =
                    *(const uint2*)&A2b[(size_t)(node0 + r) * K2 + (k0 + k)];
            }
        }
        // ---- stage B tile (256 cols x 32 k, 16 KB) ----
        const unsigned short* Bsrc = Btg + (size_t)s * 8192;
#if USE_ASYNC_LDS
        for (int i = tid; i < 1024; i += 128) {
            __builtin_amdgcn_global_load_async_to_lds_b128(
                (g_v4i_ptr)(Bsrc + i * 8),
                (l_v4i_ptr)(&Bt[i * 8]),
                0, 0);
        }
#  if __has_builtin(__builtin_amdgcn_s_wait_asynccnt)
        __builtin_amdgcn_s_wait_asynccnt(0);
#  else
        asm volatile("s_wait_asynccnt 0x0" ::: "memory");
#  endif
#else
        for (int i = tid; i < 1024; i += 128)
            ((uint4*)Bt)[i] = ((const uint4*)Bsrc)[i];
#endif
        __syncthreads();

        // ---- fragments (contiguous k = 16*hi + e in A and B) + 4 WMMAs ----
        u16x16 au;
#pragma unroll
        for (int e = 0; e < 16; ++e) {
            int k = hi * 16 + e;
            au[e] = As[row * 32 + k];
        }
        v16bf a = __builtin_bit_cast(v16bf, au);

#pragma unroll
        for (int c = 0; c < 4; ++c) {
            int col = wave * 64 + c * 16 + row;      // N index = lane & 15
            u16x16 bu;
#pragma unroll
            for (int e = 0; e < 16; ++e)             // K = 16*hi + e
                bu[e] = Bt[col * 32 + hi * 16 + e];
            v16bf b = __builtin_bit_cast(v16bf, bu);
            acc[c] = __builtin_amdgcn_wmma_f32_16x16x32_bf16(
                false, a, false, b, (short)0, acc[c], false, false);
        }
    }

    // ---- epilogue: bias + leaky, then store-bf16 or blend+pool ----
#pragma unroll
    for (int c = 0; c < 4; ++c) {
        int col = wave * 64 + c * 16 + row;
        float bcol = bias[col];
#pragma unroll
        for (int r = 0; r < 8; ++r) {
            int m = r + hi * 8;                      // C/D: M = r + 8*hi
            int node = node0 + m;
            float v = acc[c][r] + bcol;
            v = (v >= 0.0f) ? v : 0.2f * v;          // leaky 0.2
            if (mode == 0) {
                out_hb[(size_t)node * 256 + col] = f2bf(v);
            } else {
                int g = batch[node];
                bool large = (sizes[g] >= 40.0f);
                float hv = bf2f(h_prevb[(size_t)node * 256 + col]);
                float blended = large ? v : hv;
                atomicAdd(&pooled[(size_t)g * 256 + col], blended);
            }
        }
    }
}

// ---------------------------------------------------------------------------
// 7) pooled /= max(sizes, 1)
// ---------------------------------------------------------------------------
__global__ void pool_div_kernel(float* __restrict__ pooled,
                                const float* __restrict__ sizes, int n) {
    int i = blockIdx.x * blockDim.x + threadIdx.x;
    if (i < n) pooled[i] /= fmaxf(sizes[i >> 8], 1.0f);
}

// ---------------------------------------------------------------------------
// 8) head part 1: z = leaky(BN(pooled @ W_lin1 + b_lin1))   [512, 32]
// ---------------------------------------------------------------------------
__global__ void head1_kernel(const float* __restrict__ pooled,
                             const float* __restrict__ W1,   // [256, 32]
                             const float* __restrict__ b1,   // [32]
                             const float* __restrict__ gamma,
                             const float* __restrict__ beta,
                             const float* __restrict__ mean,
                             const float* __restrict__ var,
                             float* __restrict__ z) {
    int i = blockIdx.x * blockDim.x + threadIdx.x;
    if (i >= N_GRAPHS * D_MID) return;
    int m = i & (D_MID - 1);
    int g = i / D_MID;
    const float* p = pooled + (size_t)g * D_H;
    float acc = 0.0f;
    for (int k = 0; k < D_H; ++k) acc += p[k] * W1[k * D_MID + m];
    acc += b1[m];
    acc = (acc - mean[m]) * rsqrtf(var[m] + 1e-5f) * gamma[m] + beta[m];
    acc = (acc >= 0.0f) ? acc : 0.2f * acc;
    z[i] = acc;
}

// ---------------------------------------------------------------------------
// 9) head part 2: out = z @ W_lin2 + b_lin2   [512, 2]
// ---------------------------------------------------------------------------
__global__ void head2_kernel(const float* __restrict__ z,
                             const float* __restrict__ W2,   // [32, 2]
                             const float* __restrict__ b2,   // [2]
                             float* __restrict__ out) {
    int i = blockIdx.x * blockDim.x + threadIdx.x;
    if (i >= N_GRAPHS * N_CLS) return;
    int k = i & (N_CLS - 1);
    int g = i / N_CLS;
    const float* zr = z + (size_t)g * D_MID;
    float acc = 0.0f;
    for (int j = 0; j < D_MID; ++j) acc += zr[j] * W2[j * N_CLS + k];
    out[i] = acc + b2[k];
}

// ---------------------------------------------------------------------------
// launch
// ---------------------------------------------------------------------------
extern "C" void kernel_launch(void* const* d_in, const int* in_sizes, int n_in,
                              void* d_out, int out_size, void* d_ws, size_t ws_size,
                              hipStream_t stream) {
    const float* x      = (const float*)d_in[0];
    const int*   eidx   = (const int*)d_in[1];
    const int*   batch  = (const int*)d_in[2];
    const float* W1_l   = (const float*)d_in[3];
    const float* b1_l   = (const float*)d_in[4];
    const float* W1_r   = (const float*)d_in[5];
    const float* W2_l   = (const float*)d_in[6];
    const float* b2_l   = (const float*)d_in[7];
    const float* W2_r   = (const float*)d_in[8];
    const float* W_lin1 = (const float*)d_in[9];
    const float* b_lin1 = (const float*)d_in[10];
    const float* bn_g   = (const float*)d_in[11];
    const float* bn_b   = (const float*)d_in[12];
    const float* bn_m   = (const float*)d_in[13];
    const float* bn_v   = (const float*)d_in[14];
    const float* W_lin2 = (const float*)d_in[15];
    const float* b_lin2 = (const float*)d_in[16];

    const int E = in_sizes[1] / 2;               // 3,200,000
    const int* src = eidx;
    const int* dst = eidx + E;

    // workspace layout (bytes); zeroed region first (contiguous)
    char* ws = (char*)d_ws;
    auto alignB = [](size_t v) { return (v + 255) & ~(size_t)255; };
    size_t off = 0;
    size_t o_deg    = off; off = alignB(off + (size_t)N_NODES * 4);
    size_t o_sizes  = off; off = alignB(off + (size_t)N_GRAPHS * 4);
    size_t o_agg1   = off; off = alignB(off + (size_t)N_NODES * D_IN * 4);
    size_t o_agg2   = off; off = alignB(off + (size_t)N_NODES * D_H * 4);
    size_t o_pooled = off; off = alignB(off + (size_t)N_GRAPHS * D_H * 4);
    size_t zeroB    = off;                         // bytes to zero
    size_t o_xb     = off; off = alignB(off + (size_t)N_NODES * D_IN * 2);
    size_t o_hb     = off; off = alignB(off + (size_t)N_NODES * D_H * 2);
    size_t o_Btg1   = off; off = alignB(off + (size_t)8  * 8192 * 2);
    size_t o_Btg2   = off; off = alignB(off + (size_t)16 * 8192 * 2);
    size_t o_z      = off; off = alignB(off + (size_t)N_GRAPHS * D_MID * 4);
    (void)ws_size; (void)n_in; (void)out_size;

    float*          deg    = (float*)(ws + o_deg);
    float*          sizes  = (float*)(ws + o_sizes);
    float*          agg1   = (float*)(ws + o_agg1);
    float*          agg2   = (float*)(ws + o_agg2);
    float*          pooled = (float*)(ws + o_pooled);
    unsigned short* xb     = (unsigned short*)(ws + o_xb);
    unsigned short* hb     = (unsigned short*)(ws + o_hb);
    unsigned short* Btg1   = (unsigned short*)(ws + o_Btg1);
    unsigned short* Btg2   = (unsigned short*)(ws + o_Btg2);
    float*          z      = (float*)(ws + o_z);

    // 0) zero accumulators
    zero_kernel<<<4096, 256, 0, stream>>>((float4*)ws, (long long)(zeroB / 16));
    // 1) pack x -> bf16 ; weights -> bf16 K-tiled layout
    pack_bf16_kernel<<<(N_NODES * D_IN + 255) / 256, 256, 0, stream>>>(x, xb, N_NODES * D_IN);
    pack_B_kernel<<<(8 * 8192 + 255) / 256, 256, 0, stream>>>(W1_l, W1_r, D_IN, Btg1);
    pack_B_kernel<<<(16 * 8192 + 255) / 256, 256, 0, stream>>>(W2_l, W2_r, D_H, Btg2);
    // 2) conv1 scatter: agg1 += x[src], deg += 1
    {
        unsigned int blocks = ((unsigned int)E * 32u + 255u) / 256u;
        scatter_f32_kernel<<<blocks, 256, 0, stream>>>(x, src, dst, agg1, deg, E);
    }
    // 3) graph sizes
    sizes_kernel<<<(N_NODES + 255) / 256, 256, 0, stream>>>(batch, sizes, N_NODES);
    // 4) conv1 GEMM: hb = bf16(leaky(agg1/deg @ W1_l + b1_l + x @ W1_r))
    sage_gemm_kernel<<<N_NODES / 16, 128, 0, stream>>>(
        agg1, xb, Btg1, b1_l, deg,
        hb, nullptr, nullptr, nullptr, nullptr,
        D_IN, D_IN, 0);
    // 5) conv2 scatter: agg2 += h[src] (bf16 gather, f32 accumulate)
    {
        unsigned int blocks = ((unsigned int)E * 32u + 255u) / 256u;
        scatter_bf16_kernel<<<blocks, 256, 0, stream>>>(hb, src, dst, agg2, E);
    }
    // 6) conv2 GEMM + blend + pool accumulate
    sage_gemm_kernel<<<N_NODES / 16, 128, 0, stream>>>(
        agg2, hb, Btg2, b2_l, deg,
        nullptr, hb, batch, sizes, pooled,
        D_H, D_H, 1);
    // 7) pooled /= max(sizes, 1)
    pool_div_kernel<<<(N_GRAPHS * D_H + 255) / 256, 256, 0, stream>>>(pooled, sizes, N_GRAPHS * D_H);
    // 8) head
    head1_kernel<<<(N_GRAPHS * D_MID + 255) / 256, 256, 0, stream>>>(
        pooled, W_lin1, b_lin1, bn_g, bn_b, bn_m, bn_v, z);
    head2_kernel<<<(N_GRAPHS * N_CLS + 255) / 256, 256, 0, stream>>>(
        z, W_lin2, b_lin2, (float*)d_out);
}
